// MyMSA_89464168776033
// MI455X (gfx1250) — matrix operand order repaired and checked by reading, not compiled, not run on hardware
//
#include <hip/hip_runtime.h>

typedef __attribute__((ext_vector_type(16))) __bf16 v16bf;
typedef __attribute__((ext_vector_type(8)))  float  v8f;
typedef unsigned int u32x4 __attribute__((ext_vector_type(4)));
typedef int          i32x4 __attribute__((ext_vector_type(4)));
typedef int          i32x8 __attribute__((ext_vector_type(8)));

union FragBF { unsigned int u[8]; v16bf v; };

__device__ __forceinline__ unsigned int pack2_bf16(float a, float b) {
  unsigned int ua = __float_as_uint(a);
  unsigned int ub = __float_as_uint(b);
  ua = (ua + 0x7FFFu + ((ua >> 16) & 1u)) >> 16;   // RNE truncation to bf16
  ub = (ub + 0x7FFFu + ((ub >> 16) & 1u)) >> 16;
  return (ua & 0xFFFFu) | (ub << 16);
}
__device__ __forceinline__ unsigned short f2bf(float a) {
  unsigned int u = __float_as_uint(a);
  return (unsigned short)((u + 0x7FFFu + ((u >> 16) & 1u)) >> 16);
}

// ---------------------------------------------------------------------------
// TDM: async 2D bf16 tile load global -> LDS (D# per ISA 08_async_tensor §8).
// tensor dims == tile dims (no OOB), data_size=2B, per-row LDS padding.
// pad_interval code: row bytes = 2^(code+1)*4 -> 64B rows: 3 (stride 17 dw),
// 128B rows: 4 (stride 33 dw), 256B rows: 5 (stride 65 dw). pad_amount 0=1dw.
// ---------------------------------------------------------------------------
__device__ __forceinline__ void tdm_load_2d_bf16(
    unsigned lds_addr, const unsigned short* gptr,
    unsigned tile_d0, unsigned tile_d1,
    unsigned long long stride0, unsigned pad_code)
{
  unsigned long long ga = (unsigned long long)gptr;
  u32x4 g0;
  g0[0] = 1u;                                        // count=1, no gather
  g0[1] = lds_addr;                                  // LDS byte address
  g0[2] = (unsigned)ga;                              // global_addr[31:0]
  g0[3] = (unsigned)((ga >> 32) & 0x1FFFFFFu) | (2u << 30);  // [56:32] | type=2
  i32x8 g1;
  g1[0] = (int)((1u << 16) | (1u << 20) | (pad_code << 22)); // 2B, pad_en, pad_interval, pad_amount=1dw
  g1[1] = (int)(tile_d0 << 16);                      // tensor_dim0[15:0]
  g1[2] = (int)(tile_d1 << 16);                      // tensor_dim0 hi=0 | tensor_dim1 lo
  g1[3] = (int)(tile_d0 << 16);                      // tensor_dim1 hi=0 | tile_dim0
  g1[4] = (int)(tile_d1 & 0xFFFFu);                  // tile_dim1 | tile_dim2=0
  g1[5] = (int)(unsigned)(stride0 & 0xFFFFFFFFull);  // tensor_dim0_stride lo
  g1[6] = (int)(unsigned)((stride0 >> 32) & 0xFFFFull);
  g1[7] = 0;
  i32x4 z4 = (i32x4)0;
#if __clang_major__ >= 23
  i32x8 z8 = (i32x8)0;
  __builtin_amdgcn_tensor_load_to_lds(g0, g1, z4, z4, z8, 0);
#else
  __builtin_amdgcn_tensor_load_to_lds(g0, g1, z4, z4, 0);
#endif
}

__device__ __forceinline__ unsigned lds_off(const void* p) {
  return (unsigned)(unsigned long long)p;            // AS3 aperture: low 32 bits = LDS offset
}

// ---------------------------------------------------------------------------
// fp32 -> bf16 bulk convert (n multiple of 4)
// ---------------------------------------------------------------------------
__global__ __launch_bounds__(256) void cvt_bf16(
    const float* __restrict__ in, unsigned short* __restrict__ out, long n4)
{
  long i = (long)blockIdx.x * blockDim.x + threadIdx.x;
  if (i < n4) {
    float4 f = reinterpret_cast<const float4*>(in)[i];
    uint2 u;
    u.x = pack2_bf16(f.x, f.y);
    u.y = pack2_bf16(f.z, f.w);
    reinterpret_cast<uint2*>(out)[i] = u;
  }
}

// ---------------------------------------------------------------------------
// C[M,N] = A[M,K]bf16 @ B[K,N]bf16 + bias.  Block tile 128x128, K-step 32,
// 8 waves.  Double-buffered TDM staging: wave 0 issues tile k+1 while the
// block computes tile k; s_wait_tensorcnt<=2 retires the older tile.
//  sA: [row][kpair] stride 17 dw (TDM pad)   -> A-frag dword reads
//  sB: [k][n] rows of 128 bf16, stride 65 dw -> B-frag via 2x b16 reads
// ---------------------------------------------------------------------------
template <bool OUT_BF16>
__global__ __launch_bounds__(256) void gemm_bf16(
    const unsigned short* __restrict__ A, int lda,
    const unsigned short* __restrict__ B, int ldb,
    const float* __restrict__ bias,
    void* __restrict__ Cp, int ldc, int K)
{
  __shared__ unsigned int sA[2][128 * 17];
  __shared__ unsigned int sB[2][32 * 65];
  const int t    = threadIdx.x;
  const int lane = t & 31, wave = t >> 5;
  const int hi   = lane >> 4, l16 = lane & 15;
  const int wm   = wave & 3,  wn  = wave >> 2;
  const long bm  = (long)blockIdx.x * 128;
  const long bn  = (long)blockIdx.y * 128;

  v8f acc[2][4];
  #pragma unroll
  for (int i = 0; i < 2; ++i)
    #pragma unroll
    for (int j = 0; j < 4; ++j) acc[i][j] = {};

  if (t == 0) { sA[0][0] = 0; sA[1][0] = 0; sB[0][0] = 0; sB[1][0] = 0; }
  __syncthreads();

  const int nk = K / 32;
  if (t < 32) {
    tdm_load_2d_bf16(lds_off(&sA[0][0]), A + bm * lda, 32, 128, (unsigned long long)lda, 3);
    tdm_load_2d_bf16(lds_off(&sB[0][0]), B + bn,       128, 32, (unsigned long long)ldb, 5);
  }

  for (int ks = 0; ks < nk; ++ks) {
    if (t < 32) {
      if (ks + 1 < nk) {
        int buf = (ks + 1) & 1;
        tdm_load_2d_bf16(lds_off(&sA[buf][0]), A + bm * lda + (ks + 1) * 32,
                         32, 128, (unsigned long long)lda, 3);
        tdm_load_2d_bf16(lds_off(&sB[buf][0]), B + (long)(ks + 1) * 32 * ldb + bn,
                         128, 32, (unsigned long long)ldb, 5);
        __builtin_amdgcn_s_wait_tensorcnt(2);
      } else {
        __builtin_amdgcn_s_wait_tensorcnt(0);
      }
    }
    __syncthreads();

    const unsigned int*   sa = &sA[ks & 1][0];
    const unsigned short* sb = reinterpret_cast<const unsigned short*>(&sB[ks & 1][0]);

    FragBF fb[4];
    #pragma unroll
    for (int nt = 0; nt < 4; ++nt) {
      int col = wn * 64 + nt * 16 + l16;
      #pragma unroll
      for (int j = 0; j < 8; ++j) {
        int k = 2 * j + ((j >> 2) << 3) + (hi << 3);
        fb[nt].u[j] = (unsigned)sb[k * 130 + col] |
                      ((unsigned)sb[(k + 1) * 130 + col] << 16);
      }
    }
    #pragma unroll
    for (int mt = 0; mt < 2; ++mt) {
      FragBF fa;
      int row = wm * 32 + mt * 16 + l16;
      #pragma unroll
      for (int v = 0; v < 8; ++v)
        fa.u[v] = sa[row * 17 + v + ((v >> 2) << 2) + (hi << 2)];
      #pragma unroll
      for (int nt = 0; nt < 4; ++nt)
        acc[mt][nt] = __builtin_amdgcn_wmma_f32_16x16x32_bf16(
            false, fa.v, false, fb[nt].v, (short)0, acc[mt][nt], false, false);
    }
    __syncthreads();
  }

  #pragma unroll
  for (int mt = 0; mt < 2; ++mt)
    #pragma unroll
    for (int nt = 0; nt < 4; ++nt)
      #pragma unroll
      for (int r = 0; r < 8; ++r) {
        long row = bm + wm * 32 + mt * 16 + r + hi * 8;
        long col = bn + wn * 64 + nt * 16 + l16;
        float v = acc[mt][nt][r] + bias[col];
        if (OUT_BF16)
          reinterpret_cast<unsigned short*>(Cp)[row * (long)ldc + col] = f2bf(v);
        else
          reinterpret_cast<float*>(Cp)[row * (long)ldc + col] = v;
      }
}

// ---------------------------------------------------------------------------
// Flash attention over bf16 qkv [B,N,3,H,DH] (row stride 3072), DH=64.
// Block = (b, h, 64-row tile), 4 waves.  Q and double-buffered K/V tiles are
// staged by TDM in natural row-major layout (rows 128B -> padded 33-dw
// stride); wave 0 prefetches key-block kb+1 while the block computes kb.
//  sQ/sK: [row][dpair]  -> A-frag / B-frag dword reads
//  sV:    [kv][d]       -> B-frag via 2x b16 reads (kv pairs along rows)
// Writes bf16 [B,N,H*DH] for the output-projection GEMM.
// ---------------------------------------------------------------------------
__global__ __launch_bounds__(128) void flash_attn(
    const unsigned short* __restrict__ qkv, unsigned short* __restrict__ out)
{
  constexpr int Nn = 2048, DH = 64, QLD = 3072, D = 1024;
  constexpr int NKB = Nn / 64;
  __shared__ unsigned int sQ[64 * 33];
  __shared__ unsigned int sK[2][64 * 33];
  __shared__ unsigned int sV[2][64 * 33];
  __shared__ unsigned int sP[4 * 16 * 33];

  const int t    = threadIdx.x;
  const int lane = t & 31, w = t >> 5;
  const int hi   = lane >> 4, l16 = lane & 15;
  const int rb   = blockIdx.x;
  const int h    = blockIdx.y;
  const int b    = blockIdx.z;
  const long headoff  = (long)h * DH;
  const long batchoff = (long)b * Nn;

  const unsigned short* qbase = qkv + (batchoff + rb * 64) * (long)QLD + headoff;
  const unsigned short* kbase = qkv + batchoff * (long)QLD + D + headoff;
  const unsigned short* vbase = qkv + batchoff * (long)QLD + 2 * D + headoff;

  if (t == 0) { sQ[0] = 0; sK[0][0] = 0; sK[1][0] = 0; sV[0][0] = 0; sV[1][0] = 0; }
  __syncthreads();

  if (t < 32) {            // Q then K0,V0: in-order TDM -> wait<=2 covers Q too
    tdm_load_2d_bf16(lds_off(&sQ[0]),    qbase, 64, 64, (unsigned long long)QLD, 4);
    tdm_load_2d_bf16(lds_off(&sK[0][0]), kbase, 64, 64, (unsigned long long)QLD, 4);
    tdm_load_2d_bf16(lds_off(&sV[0][0]), vbase, 64, 64, (unsigned long long)QLD, 4);
  }

  FragBF aq[2];
  float mrow[8], lrow[8];
  v8f o[4];
  #pragma unroll
  for (int r = 0; r < 8; ++r) { mrow[r] = -3.0e38f; lrow[r] = 0.f; }
  #pragma unroll
  for (int nt = 0; nt < 4; ++nt) o[nt] = {};

  for (int kb = 0; kb < NKB; ++kb) {
    if (t < 32) {
      if (kb + 1 < NKB) {
        int buf = (kb + 1) & 1;
        tdm_load_2d_bf16(lds_off(&sK[buf][0]), kbase + (long)(kb + 1) * 64 * QLD,
                         64, 64, (unsigned long long)QLD, 4);
        tdm_load_2d_bf16(lds_off(&sV[buf][0]), vbase + (long)(kb + 1) * 64 * QLD,
                         64, 64, (unsigned long long)QLD, 4);
        __builtin_amdgcn_s_wait_tensorcnt(2);
      } else {
        __builtin_amdgcn_s_wait_tensorcnt(0);
      }
    }
    __syncthreads();

    if (kb == 0) {         // persistent Q A-fragments (K=64 -> 2 ksteps)
      #pragma unroll
      for (int ks = 0; ks < 2; ++ks)
        #pragma unroll
        for (int v = 0; v < 8; ++v)
          aq[ks].u[v] = sQ[(w * 16 + l16) * 33 + ks * 16 + v + ((v >> 2) << 2) + (hi << 2)];
    }

    const unsigned int*   sk = &sK[kb & 1][0];
    const unsigned short* sv = reinterpret_cast<const unsigned short*>(&sV[kb & 1][0]);

    v8f s[4];
    #pragma unroll
    for (int nt = 0; nt < 4; ++nt) {         // S = Q K^T / sqrt(DH)
      s[nt] = {};
      #pragma unroll
      for (int ks = 0; ks < 2; ++ks) {
        FragBF bk;
        #pragma unroll
        for (int j = 0; j < 8; ++j)
          bk.u[j] = sk[(nt * 16 + l16) * 33 + ks * 16 + j + ((j >> 2) << 2) + (hi << 2)];
        s[nt] = __builtin_amdgcn_wmma_f32_16x16x32_bf16(
            false, aq[ks].v, false, bk.v, (short)0, s[nt], false, false);
      }
      s[nt] = s[nt] * 0.125f;
    }

    unsigned short* pw = reinterpret_cast<unsigned short*>(&sP[w * 528]);
    #pragma unroll
    for (int r = 0; r < 8; ++r) {            // streaming softmax
      float mx = fmaxf(fmaxf(s[0][r], s[1][r]), fmaxf(s[2][r], s[3][r]));
      #pragma unroll
      for (int msk = 8; msk >= 1; msk >>= 1) mx = fmaxf(mx, __shfl_xor(mx, msk, 16));
      float nm = fmaxf(mrow[r], mx);
      float al = __expf(mrow[r] - nm);
      float rs = 0.f;
      #pragma unroll
      for (int nt = 0; nt < 4; ++nt) {
        float pv = __expf(s[nt][r] - nm);
        s[nt][r] = pv;
        rs += pv;
      }
      #pragma unroll
      for (int msk = 8; msk >= 1; msk >>= 1) rs += __shfl_xor(rs, msk, 16);
      lrow[r] = lrow[r] * al + rs;
      mrow[r] = nm;
      #pragma unroll
      for (int nt = 0; nt < 4; ++nt) o[nt][r] *= al;
      #pragma unroll
      for (int nt = 0; nt < 4; ++nt)
        pw[(r + 8 * hi) * 66 + nt * 16 + l16] = f2bf(s[nt][r]);
    }
    __syncthreads();       // P visible for A-frag reads

    FragBF ap[2];
    #pragma unroll
    for (int ks = 0; ks < 2; ++ks)
      #pragma unroll
      for (int v = 0; v < 8; ++v)
        ap[ks].u[v] = sP[w * 528 + l16 * 33 + ks * 16 + v + ((v >> 2) << 2) + (hi << 2)];
    #pragma unroll
    for (int nt = 0; nt < 4; ++nt)           // O += P V
      #pragma unroll
      for (int ks = 0; ks < 2; ++ks) {
        FragBF bv;
        #pragma unroll
        for (int j = 0; j < 8; ++j) {
          int kv = ks * 32 + 2 * j + ((j >> 2) << 3) + (hi << 3);
          int d  = nt * 16 + l16;
          bv.u[j] = (unsigned)sv[kv * 66 + d] |
                    ((unsigned)sv[(kv + 1) * 66 + d] << 16);
        }
        o[nt] = __builtin_amdgcn_wmma_f32_16x16x32_bf16(
            false, ap[ks].v, false, bv.v, (short)0, o[nt], false, false);
      }
    __syncthreads();       // all sK/sV/sP reads done before next TDM overwrite
  }

  #pragma unroll
  for (int nt = 0; nt < 4; ++nt)
    #pragma unroll
    for (int r = 0; r < 8; ++r) {
      long row = batchoff + rb * 64 + w * 16 + r + 8 * hi;
      long col = headoff + nt * 16 + l16;
      out[row * (long)D + col] = f2bf(o[nt][r] / lrow[r]);
    }
}

extern "C" void kernel_launch(void* const* d_in, const int* in_sizes, int n_in,
                              void* d_out, int out_size, void* d_ws, size_t ws_size,
                              hipStream_t stream) {
  (void)in_sizes; (void)n_in; (void)out_size; (void)ws_size;
  const float* x     = (const float*)d_in[0];
  const float* W_qkv = (const float*)d_in[1];
  const float* b_qkv = (const float*)d_in[2];
  const float* W_out = (const float*)d_in[3];
  const float* b_out = (const float*)d_in[4];
  float* out = (float*)d_out;

  unsigned short* xb    = (unsigned short*)d_ws;           // 4096 x 1024
  unsigned short* wqkvb = xb    + (size_t)4096 * 1024;     // 1024 x 3072
  unsigned short* woutb = wqkvb + (size_t)1024 * 3072;     // 1024 x 1024
  unsigned short* qkvb  = woutb + (size_t)1024 * 1024;     // 4096 x 3072
  unsigned short* attnb = qkvb  + (size_t)4096 * 3072;     // 4096 x 1024

  cvt_bf16<<<4096, 256, 0, stream>>>(x,     xb,    (long)4096 * 1024 / 4);
  cvt_bf16<<<3072, 256, 0, stream>>>(W_qkv, wqkvb, (long)1024 * 3072 / 4);
  cvt_bf16<<<1024, 256, 0, stream>>>(W_out, woutb, (long)1024 * 1024 / 4);

  dim3 g1(4096 / 128, 3072 / 128);
  gemm_bf16<true><<<g1, 256, 0, stream>>>(xb, 1024, wqkvb, 3072, b_qkv, qkvb, 3072, 1024);

  dim3 g2(2048 / 64, 16, 2);
  flash_attn<<<g2, 128, 0, stream>>>(qkvb, attnb);

  dim3 g3(4096 / 128, 1024 / 128);
  gemm_bf16<false><<<g3, 256, 0, stream>>>(attnb, 1024, woutb, 1024, b_out, out, 1024, 1024);
}
